// ASPPDeformable_65824668778876
// MI455X (gfx1250) — matrix operand order repaired and checked by reading, not compiled, not run on hardware
//
#include <hip/hip_runtime.h>
#include <hip/hip_bf16.h>
#include <math.h>

#define BATCH  2
#define HH     64
#define WWID   64
#define CINCH  256
#define MTOT   (BATCH*HH*WWID)   /* 8192 pixels */
#define PLANES 256
#define CATC   1280
#define EPSBN  1e-5f

typedef __bf16 bf16_t;
typedef bf16_t v16bf __attribute__((ext_vector_type(16)));
typedef float  v8f   __attribute__((ext_vector_type(8)));
typedef unsigned int u32x4 __attribute__((ext_vector_type(4)));
typedef int          i32x4 __attribute__((ext_vector_type(4)));
typedef int          i32x8 __attribute__((ext_vector_type(8)));

union Frag16 { v16bf v; unsigned short s[16]; uint4 q[2]; };

static __device__ __forceinline__ unsigned short f2bf(float f) {
  union { float f; unsigned u; } v; v.f = f;
  unsigned u = v.u;
  unsigned r = u + 0x7FFFu + ((u >> 16) & 1u);   // round-to-nearest-even
  return (unsigned short)(r >> 16);
}
static __device__ __forceinline__ float bf2f(unsigned short h) {
  union { unsigned u; float f; } v; v.u = ((unsigned)h) << 16;
  return v.f;
}

// ---------------------------------------------------------------------------
// Tensor Data Mover: DMA a fully-in-bounds 2D bf16 tile (tile_rows x row_elems,
// row stride = row_elems) from global into LDS.  D# per CDNA5 ISA §8.3/8.4.
// ---------------------------------------------------------------------------
static __device__ __forceinline__ void tdm_load_tile(unsigned lds_off,
                                                     const unsigned short* gptr,
                                                     unsigned row_elems,
                                                     unsigned tile_rows) {
  unsigned long long ga = (unsigned long long)(size_t)gptr;
  u32x4 g0;
  g0[0] = 1u;                                            // count=1, user mode
  g0[1] = lds_off;                                       // lds_addr (bytes)
  g0[2] = (unsigned)(ga & 0xFFFFFFFFull);                // global_addr lo
  g0[3] = (unsigned)((ga >> 32) & 0x1FFFFFFull) | (2u << 30); // addr hi | type=2
  i32x8 g1;
  g1[0] = 0x10000;                                       // data_size=1 (2 bytes)
  g1[1] = (int)(row_elems << 16);                        // tensor_dim0[15:0] @63:48
  g1[2] = (int)((tile_rows & 0xFFFFu) << 16);            // dim0 hi=0 | tensor_dim1 lo
  g1[3] = (int)(((tile_rows >> 16) & 0xFFFFu) | (row_elems << 16)); // dim1 hi | tile_dim0
  g1[4] = (int)tile_rows;                                // tile_dim1 (tile_dim2=0)
  g1[5] = (int)row_elems;                                // tensor_dim0_stride lo32
  g1[6] = 0;
  g1[7] = 0;
  i32x4 z4 = {0, 0, 0, 0};
#if defined(__clang_major__) && (__clang_major__ >= 23)
  i32x8 z8 = {0, 0, 0, 0, 0, 0, 0, 0};
  __builtin_amdgcn_tensor_load_to_lds(g0, g1, z4, z4, z8, 0);
#else
  __builtin_amdgcn_tensor_load_to_lds(g0, g1, z4, z4, 0);
#endif
}

// ---------------------------------------------------------------------------
// x: NCHW f32 -> NHWC bf16
// ---------------------------------------------------------------------------
__global__ void k_convert_x(const float* __restrict__ x, unsigned short* __restrict__ xh) {
  int idx = blockIdx.x * blockDim.x + threadIdx.x;
  if (idx >= MTOT * CINCH) return;
  int c   = idx & (CINCH - 1);
  int m   = idx >> 8;
  int b   = m >> 12;
  int pix = m & 4095;
  xh[idx] = f2bf(x[((size_t)(b * CINCH + c) << 12) + pix]);
}

// ---------------------------------------------------------------------------
// Weights: OIHW f32 -> [K = tap*C + c][Npad] bf16, zero-padded in N
// ---------------------------------------------------------------------------
__global__ void k_prep_w(const float* __restrict__ src, unsigned short* __restrict__ dst,
                         int O, int C, int KK, int Npad) {
  int idx  = blockIdx.x * blockDim.x + threadIdx.x;
  int Ktot = KK * C;
  if (idx >= Ktot * Npad) return;
  int n   = idx % Npad;
  int k   = idx / Npad;
  int tap = k / C;
  int c   = k - tap * C;
  float v = 0.f;
  if (n < O) v = src[((size_t)n * C + c) * KK + tap];
  dst[idx] = f2bf(v);
}

// ---------------------------------------------------------------------------
// Implicit-GEMM conv (offset / modulation convs), M-tile = 32 pixels,
// double-buffered LDS A tiles (one barrier per 32-K step), Npad <= 128:
// wave w owns N tile [w*16, w*16+16), 2 M-subtiles -> 2 WMMAs per step.
// KKW/PADW compile-time: no integer-division chains in the hot loop.
// epilogue 0: +bias -> f32 ; 1: 2*sigmoid(v+bias) -> f32
// ---------------------------------------------------------------------------
template <int KKW, int PADW>
__global__ void k_conv_gemm(const unsigned short* __restrict__ xh,
                            const unsigned short* __restrict__ Bm,
                            const float* __restrict__ bias,
                            float* __restrict__ out,
                            int Cout, int Npad, int epilogue) {
  constexpr int KK = KKW * KKW;
  constexpr int S  = KK * 8;
  __shared__ __attribute__((aligned(16))) unsigned short abuf[2][32 * 32];
  const int t      = threadIdx.x;
  const int lane   = t & 31;
  const int wave   = t >> 5;
  const int lane16 = lane & 15;
  const int hi     = lane >> 4;
  const int mbase  = blockIdx.x * 32;
  const int ntile  = wave * 16;
  const bool active = ntile < Npad;

  v8f acc0 = {0.f, 0.f, 0.f, 0.f, 0.f, 0.f, 0.f, 0.f};
  v8f acc1 = {0.f, 0.f, 0.f, 0.f, 0.f, 0.f, 0.f, 0.f};

  auto stageA = [&](int sn) {
    int tap = sn >> 3;
    int cb  = (sn & 7) * 32;
    int di  = tap / KKW - PADW;      // compile-time strength-reduced
    int dj  = tap % KKW - PADW;
    int p   = t >> 3;
    int c4  = (t & 7) * 4;
    int m   = mbase + p;
    int b   = m >> 12;
    int pix = m & 4095;
    int h = pix >> 6, w = pix & 63;
    int yy = h + di, xx = w + dj;
    uint2 val; val.x = 0u; val.y = 0u;
    if (yy >= 0 && yy < HH && xx >= 0 && xx < WWID)
      val = *(const uint2*)&xh[(((size_t)(b * HH + yy) * WWID + xx) << 8) + cb + c4];
    *(uint2*)&abuf[sn & 1][p * 32 + c4] = val;
  };

  stageA(0);
  for (int s = 0; s < S; ++s) {
    __syncthreads();
    if (active) {
      const unsigned short* ab = &abuf[s & 1][0];
      Frag16 a0, a1, bf;
      a0.q[0] = *(const uint4*)&ab[lane16 * 32 + hi * 8];
      a0.q[1] = *(const uint4*)&ab[lane16 * 32 + 16 + hi * 8];
      a1.q[0] = *(const uint4*)&ab[(lane16 + 16) * 32 + hi * 8];
      a1.q[1] = *(const uint4*)&ab[(lane16 + 16) * 32 + 16 + hi * 8];
      int kb  = (s >> 3) * CINCH + (s & 7) * 32;
      const unsigned short* brow = &Bm[(size_t)(kb + hi * 8) * Npad + ntile + lane16];
      for (int i = 0; i < 8; ++i) {
        bf.s[i]     = brow[(size_t)i * Npad];
        bf.s[8 + i] = brow[(size_t)(16 + i) * Npad];
      }
      acc0 = __builtin_amdgcn_wmma_f32_16x16x32_bf16(false, a0.v, false, bf.v,
                                                     (short)0, acc0, false, false);
      acc1 = __builtin_amdgcn_wmma_f32_16x16x32_bf16(false, a1.v, false, bf.v,
                                                     (short)0, acc1, false, false);
    }
    if (s + 1 < S) stageA(s + 1);
  }
  if (active) {
    int n = ntile + lane16;
    if (n < Cout) {
      float bv = bias[n];
      for (int r = 0; r < 8; ++r) {
        int m0 = mbase + r + 8 * hi;
        float v0 = acc0[r] + bv;
        float v1 = acc1[r] + bv;
        if (epilogue == 1) {
          v0 = 2.0f / (1.0f + __expf(-v0));
          v1 = 2.0f / (1.0f + __expf(-v1));
        }
        out[(size_t)m0 * Cout + n]        = v0;
        out[(size_t)(m0 + 16) * Cout + n] = v1;
      }
    }
  }
}

// ---------------------------------------------------------------------------
// Deformable conv main GEMM, M-tile = 32 pixels, N = 256 (2 tiles/wave),
// 4 WMMAs per 32-K step.  A = bilinear-sampled im2col tile (LDS, dbl-buffered);
// B = weight panel DMA'd by the Tensor Data Mover into LDS (dbl-buffered),
// overlapped with compute and synchronized via s_wait_tensorcnt + barrier.
// Epilogue: BN + ReLU -> bf16 into concat buffer at channel offset choff.
// ---------------------------------------------------------------------------
template <int KKW, int PADW>
__global__ void k_deform_gemm(const unsigned short* __restrict__ xh,
                              const float* __restrict__ off,
                              const float* __restrict__ mod,
                              const unsigned short* __restrict__ Wm,   // [KK*256][256]
                              const float* __restrict__ bng, const float* __restrict__ bnb,
                              const float* __restrict__ bnm, const float* __restrict__ bnv,
                              unsigned short* __restrict__ cat, int choff) {
  constexpr int KK = KKW * KKW;
  constexpr int S  = KK * 8;
  __shared__ __attribute__((aligned(16))) unsigned short abuf[2][32 * 32];
  __shared__ __attribute__((aligned(16))) unsigned short bbuf[2][32 * 256];
  __shared__ int   p_y0[2][32], p_x0[2][32];
  __shared__ float p_w00[2][32], p_w01[2][32], p_w10[2][32], p_w11[2][32];

  const int t      = threadIdx.x;
  const int lane   = t & 31;
  const int wave   = t >> 5;
  const int lane16 = lane & 15;
  const int hi     = lane >> 4;
  const int mbase  = blockIdx.x * 32;
  const int n0     = wave * 32;
  const int n1     = n0 + 16;

  v8f acc00 = {0.f,0.f,0.f,0.f,0.f,0.f,0.f,0.f};
  v8f acc01 = {0.f,0.f,0.f,0.f,0.f,0.f,0.f,0.f};
  v8f acc10 = {0.f,0.f,0.f,0.f,0.f,0.f,0.f,0.f};
  v8f acc11 = {0.f,0.f,0.f,0.f,0.f,0.f,0.f,0.f};

  auto calcParams = [&](int tap) {           // threads 0..31: per-pixel tap params
    int q = tap & 1;
    int p = t;
    int m = mbase + p;
    int pix = m & 4095;
    int h = pix >> 6, w = pix & 63;
    int gi = tap / KKW, gj = tap % KKW;      // constant-divisor
    float oy = off[(size_t)m * (2 * KK) + 2 * tap + 0];
    float ox = off[(size_t)m * (2 * KK) + 2 * tap + 1];
    float mo = mod[(size_t)m * KK + tap];
    float py = (float)(h - PADW + gi) + oy;
    float px = (float)(w - PADW + gj) + ox;
    float fy = floorf(py), fx = floorf(px);
    float wy = py - fy, wx = px - fx;
    p_y0[q][p]  = (int)fy;
    p_x0[q][p]  = (int)fx;
    p_w00[q][p] = (1.f - wy) * (1.f - wx) * mo;
    p_w01[q][p] = (1.f - wy) * wx * mo;
    p_w10[q][p] = wy * (1.f - wx) * mo;
    p_w11[q][p] = wy * wx * mo;
  };

  auto stageA = [&](int sn) {                // sampled 32x32 bf16 tile, 4 elems/thread
    int q   = (sn >> 3) & 1;
    int cb  = (sn & 7) * 32;
    int ci  = t & 31;
    int pb  = t >> 5;
#pragma unroll
    for (int j = 0; j < 4; ++j) {
      int p  = pb + j * 8;
      int m  = mbase + p;
      int b  = m >> 12;
      int c  = cb + ci;
      int y0 = p_y0[q][p], x0 = p_x0[q][p];
      float w00 = p_w00[q][p], w01 = p_w01[q][p], w10 = p_w10[q][p], w11 = p_w11[q][p];
      const unsigned short* xb = xh + (((size_t)b * HH * WWID) << 8) + c;
      float v = 0.f;
      if ((unsigned)y0 < (unsigned)HH) {
        if ((unsigned)x0       < (unsigned)WWID) v += w00 * bf2f(xb[(size_t)(y0 * WWID + x0)     << 8]);
        if ((unsigned)(x0 + 1) < (unsigned)WWID) v += w01 * bf2f(xb[(size_t)(y0 * WWID + x0 + 1) << 8]);
      }
      if ((unsigned)(y0 + 1) < (unsigned)HH) {
        if ((unsigned)x0       < (unsigned)WWID) v += w10 * bf2f(xb[(size_t)((y0 + 1) * WWID + x0)     << 8]);
        if ((unsigned)(x0 + 1) < (unsigned)WWID) v += w11 * bf2f(xb[(size_t)((y0 + 1) * WWID + x0 + 1) << 8]);
      }
      abuf[sn & 1][p * 32 + ci] = f2bf(v);
    }
  };

  auto issueB = [&](int sn) {                // wave 0 only: TDM next B tile -> LDS
    int kb = (sn >> 3) * CINCH + (sn & 7) * 32;
    unsigned lds_off = (unsigned)(unsigned long long)(size_t)(void*)&bbuf[sn & 1][0];
    tdm_load_tile(lds_off, Wm + (size_t)kb * PLANES, PLANES, 32);
  };

  // prologue: params for tap 0, A(0) staged, B(0) in flight
  if (t < 32) calcParams(0);
  __syncthreads();
  stageA(0);
  if (wave == 0) issueB(0);

  for (int s = 0; s < S; ++s) {
    if (s + 1 < S && ((s + 1) & 7) == 0 && t < 32) calcParams((s + 1) >> 3);
    if (wave == 0) __builtin_amdgcn_s_wait_tensorcnt(0);   // B(s) landed
    __syncthreads();                                        // publish B(s), A(s)
    if (wave == 0 && s + 1 < S) issueB(s + 1);              // overlap next DMA
    {
      const unsigned short* ab = &abuf[s & 1][0];
      const unsigned short* bb = &bbuf[s & 1][0];
      Frag16 a0, a1, b0, b1;
      a0.q[0] = *(const uint4*)&ab[lane16 * 32 + hi * 8];
      a0.q[1] = *(const uint4*)&ab[lane16 * 32 + 16 + hi * 8];
      a1.q[0] = *(const uint4*)&ab[(lane16 + 16) * 32 + hi * 8];
      a1.q[1] = *(const uint4*)&ab[(lane16 + 16) * 32 + 16 + hi * 8];
      for (int i = 0; i < 8; ++i) {
        b0.s[i]     = bb[(hi * 8 + i) * 256 + n0 + lane16];
        b0.s[8 + i] = bb[(16 + hi * 8 + i) * 256 + n0 + lane16];
        b1.s[i]     = bb[(hi * 8 + i) * 256 + n1 + lane16];
        b1.s[8 + i] = bb[(16 + hi * 8 + i) * 256 + n1 + lane16];
      }
      acc00 = __builtin_amdgcn_wmma_f32_16x16x32_bf16(false, a0.v, false, b0.v, (short)0, acc00, false, false);
      acc01 = __builtin_amdgcn_wmma_f32_16x16x32_bf16(false, a0.v, false, b1.v, (short)0, acc01, false, false);
      acc10 = __builtin_amdgcn_wmma_f32_16x16x32_bf16(false, a1.v, false, b0.v, (short)0, acc10, false, false);
      acc11 = __builtin_amdgcn_wmma_f32_16x16x32_bf16(false, a1.v, false, b1.v, (short)0, acc11, false, false);
    }
    if (s + 1 < S) stageA(s + 1);
  }

  {                                          // BN + ReLU -> bf16 into concat
    int nA = n0 + lane16;
    int nB = n1 + lane16;
    float sA = rsqrtf(bnv[nA] + EPSBN) * bng[nA];
    float tA = bnb[nA] - bnm[nA] * sA;
    float sB = rsqrtf(bnv[nB] + EPSBN) * bng[nB];
    float tB = bnb[nB] - bnm[nB] * sB;
    for (int r = 0; r < 8; ++r) {
      int m0 = mbase + r + 8 * hi;
      int m1 = m0 + 16;
      float v;
      v = acc00[r] * sA + tA; cat[(size_t)m0 * CATC + choff + nA] = f2bf(v > 0.f ? v : 0.f);
      v = acc01[r] * sB + tB; cat[(size_t)m0 * CATC + choff + nB] = f2bf(v > 0.f ? v : 0.f);
      v = acc10[r] * sA + tA; cat[(size_t)m1 * CATC + choff + nA] = f2bf(v > 0.f ? v : 0.f);
      v = acc11[r] * sB + tB; cat[(size_t)m1 * CATC + choff + nB] = f2bf(v > 0.f ? v : 0.f);
    }
  }
}

// ---------------------------------------------------------------------------
// Pool branch
// ---------------------------------------------------------------------------
__global__ void k_pool_mean(const float* __restrict__ x, float* __restrict__ xmean) {
  __shared__ float red[256];
  int bc = blockIdx.x;
  const float* p = x + (size_t)bc * 4096;
  float s = 0.f;
  for (int i = threadIdx.x; i < 4096; i += 256) s += p[i];
  red[threadIdx.x] = s;
  __syncthreads();
  for (int st = 128; st > 0; st >>= 1) {
    if (threadIdx.x < st) red[threadIdx.x] += red[threadIdx.x + st];
    __syncthreads();
  }
  if (threadIdx.x == 0) xmean[bc] = red[0] * (1.0f / 4096.0f);
}

__global__ void k_pool_conv(const float* __restrict__ xmean, const float* __restrict__ pw,
                            const float* __restrict__ g, const float* __restrict__ b,
                            const float* __restrict__ m, const float* __restrict__ v,
                            float* __restrict__ pout) {
  int idx = blockIdx.x * blockDim.x + threadIdx.x;
  if (idx >= BATCH * PLANES) return;
  int bb = idx >> 8, oc = idx & 255;
  const float* xm = xmean + bb * CINCH;
  const float* wr = pw + (size_t)oc * CINCH;
  float s = 0.f;
  for (int c = 0; c < CINCH; ++c) s += xm[c] * wr[c];
  float sc = rsqrtf(v[oc] + EPSBN) * g[oc];
  float val = s * sc + (b[oc] - m[oc] * sc);
  pout[idx] = val > 0.f ? val : 0.f;
}

__global__ void k_pool_bcast(const float* __restrict__ pout, unsigned short* __restrict__ cat) {
  int idx = blockIdx.x * blockDim.x + threadIdx.x;
  if (idx >= MTOT * PLANES) return;
  int c = idx & 255;
  int m = idx >> 8;
  int b = m >> 12;
  cat[(size_t)m * CATC + 1024 + c] = f2bf(pout[b * PLANES + c]);
}

// ---------------------------------------------------------------------------
// Fuse 1x1 conv: M=8192, K=1280, N=256 GEMM, M-tile 32, 4 WMMAs per K-step;
// A direct from concat (bf16), BN + ReLU epilogue, f32 NCHW to d_out.
// ---------------------------------------------------------------------------
__global__ void k_fuse_gemm(const unsigned short* __restrict__ cat,
                            const unsigned short* __restrict__ Wm,   // [1280][256]
                            const float* __restrict__ g, const float* __restrict__ bb,
                            const float* __restrict__ mm, const float* __restrict__ vv,
                            float* __restrict__ out) {
  const int t      = threadIdx.x;
  const int lane   = t & 31;
  const int wave   = t >> 5;
  const int lane16 = lane & 15;
  const int hi     = lane >> 4;
  const int mbase  = blockIdx.x * 32;
  const int n0     = wave * 32;
  const int n1     = n0 + 16;

  v8f acc00 = {0.f,0.f,0.f,0.f,0.f,0.f,0.f,0.f};
  v8f acc01 = {0.f,0.f,0.f,0.f,0.f,0.f,0.f,0.f};
  v8f acc10 = {0.f,0.f,0.f,0.f,0.f,0.f,0.f,0.f};
  v8f acc11 = {0.f,0.f,0.f,0.f,0.f,0.f,0.f,0.f};

  const unsigned short* arow0 = cat + (size_t)(mbase + lane16) * CATC;
  const unsigned short* arow1 = cat + (size_t)(mbase + 16 + lane16) * CATC;
  for (int kb = 0; kb < CATC; kb += 32) {
    Frag16 a0, a1, b0, b1;
    a0.q[0] = *(const uint4*)(arow0 + kb + hi * 8);
    a0.q[1] = *(const uint4*)(arow0 + kb + 16 + hi * 8);
    a1.q[0] = *(const uint4*)(arow1 + kb + hi * 8);
    a1.q[1] = *(const uint4*)(arow1 + kb + 16 + hi * 8);
    const unsigned short* brow = &Wm[(size_t)(kb + hi * 8) * PLANES + lane16];
    __builtin_prefetch(brow + 32 * PLANES, 0, 1);
    for (int i = 0; i < 8; ++i) {
      b0.s[i]     = brow[(size_t)i * PLANES + n0];
      b0.s[8 + i] = brow[(size_t)(16 + i) * PLANES + n0];
      b1.s[i]     = brow[(size_t)i * PLANES + n1];
      b1.s[8 + i] = brow[(size_t)(16 + i) * PLANES + n1];
    }
    acc00 = __builtin_amdgcn_wmma_f32_16x16x32_bf16(false, a0.v, false, b0.v, (short)0, acc00, false, false);
    acc01 = __builtin_amdgcn_wmma_f32_16x16x32_bf16(false, a0.v, false, b1.v, (short)0, acc01, false, false);
    acc10 = __builtin_amdgcn_wmma_f32_16x16x32_bf16(false, a1.v, false, b0.v, (short)0, acc10, false, false);
    acc11 = __builtin_amdgcn_wmma_f32_16x16x32_bf16(false, a1.v, false, b1.v, (short)0, acc11, false, false);
  }
  int nA = n0 + lane16;
  int nB = n1 + lane16;
  float sA = rsqrtf(vv[nA] + EPSBN) * g[nA];
  float tA = bb[nA] - mm[nA] * sA;
  float sB = rsqrtf(vv[nB] + EPSBN) * g[nB];
  float tB = bb[nB] - mm[nB] * sB;
  for (int r = 0; r < 8; ++r) {
    int m0  = mbase + r + 8 * hi;
    int m1  = m0 + 16;
    int b0i = m0 >> 12, p0 = m0 & 4095;
    int b1i = m1 >> 12, p1 = m1 & 4095;
    float v;
    v = acc00[r] * sA + tA; out[((size_t)(b0i * PLANES + nA) << 12) + p0] = v > 0.f ? v : 0.f;
    v = acc01[r] * sB + tB; out[((size_t)(b0i * PLANES + nB) << 12) + p0] = v > 0.f ? v : 0.f;
    v = acc10[r] * sA + tA; out[((size_t)(b1i * PLANES + nA) << 12) + p1] = v > 0.f ? v : 0.f;
    v = acc11[r] * sB + tB; out[((size_t)(b1i * PLANES + nB) << 12) + p1] = v > 0.f ? v : 0.f;
  }
}

// ---------------------------------------------------------------------------
extern "C" void kernel_launch(void* const* d_in, const int* in_sizes, int n_in,
                              void* d_out, int out_size, void* d_ws, size_t ws_size,
                              hipStream_t stream) {
  (void)in_sizes; (void)n_in; (void)out_size; (void)ws_size;
  const float* x = (const float*)d_in[0];
  static const int ks[4] = {1, 1, 3, 7};

  size_t cur = 0;
  auto carve = [&](size_t bytes) -> void* {
    void* p = (char*)d_ws + cur;
    cur += (bytes + 255) & ~(size_t)255;
    return p;
  };

  unsigned short* xh  = (unsigned short*)carve((size_t)MTOT * CINCH * 2);
  unsigned short* cat = (unsigned short*)carve((size_t)MTOT * CATC * 2);
  unsigned short* wT[4];  unsigned short* offT[4]; unsigned short* modT[4];
  float* offbuf[4]; float* modbuf[4];
  int npadOff[4], npadMod[4];
  for (int i = 0; i < 4; ++i) {
    int KK = ks[i] * ks[i];
    npadOff[i] = ((2 * KK + 15) / 16) * 16;
    npadMod[i] = ((KK + 15) / 16) * 16;
    wT[i]     = (unsigned short*)carve((size_t)KK * CINCH * PLANES * 2);
    offT[i]   = (unsigned short*)carve((size_t)KK * CINCH * npadOff[i] * 2);
    modT[i]   = (unsigned short*)carve((size_t)KK * CINCH * npadMod[i] * 2);
    offbuf[i] = (float*)carve((size_t)MTOT * 2 * KK * 4);
    modbuf[i] = (float*)carve((size_t)MTOT * KK * 4);
  }
  unsigned short* fuseT = (unsigned short*)carve((size_t)CATC * PLANES * 2);
  float* xmean = (float*)carve((size_t)BATCH * CINCH * 4);
  float* pout  = (float*)carve((size_t)BATCH * PLANES * 4);

  // 1. input -> NHWC bf16
  k_convert_x<<<(MTOT * CINCH + 255) / 256, 256, 0, stream>>>(x, xh);

  // 2. weight prep
  for (int i = 0; i < 4; ++i) {
    int KK = ks[i] * ks[i];
    const float* off_w = (const float*)d_in[1 + 9 * i + 0];
    const float* mod_w = (const float*)d_in[1 + 9 * i + 2];
    const float* w     = (const float*)d_in[1 + 9 * i + 4];
    int nMain = KK * CINCH * PLANES;
    k_prep_w<<<(nMain + 255) / 256, 256, 0, stream>>>(w, wT[i], PLANES, CINCH, KK, PLANES);
    int nOff = KK * CINCH * npadOff[i];
    k_prep_w<<<(nOff + 255) / 256, 256, 0, stream>>>(off_w, offT[i], 2 * KK, CINCH, KK, npadOff[i]);
    int nMod = KK * CINCH * npadMod[i];
    k_prep_w<<<(nMod + 255) / 256, 256, 0, stream>>>(mod_w, modT[i], KK, CINCH, KK, npadMod[i]);
  }
  {
    const float* w1 = (const float*)d_in[42];
    int nF = CATC * PLANES;
    k_prep_w<<<(nF + 255) / 256, 256, 0, stream>>>(w1, fuseT, PLANES, CATC, 1, PLANES);
  }

  // 3. offset + modulation convs (WMMA implicit GEMM, templated on k)
  for (int i = 0; i < 4; ++i) {
    int KK = ks[i] * ks[i];
    const float* off_b = (const float*)d_in[1 + 9 * i + 1];
    const float* mod_b = (const float*)d_in[1 + 9 * i + 3];
    switch (ks[i]) {
      case 1:
        k_conv_gemm<1, 0><<<MTOT / 32, 256, 0, stream>>>(xh, offT[i], off_b, offbuf[i], 2 * KK, npadOff[i], 0);
        k_conv_gemm<1, 0><<<MTOT / 32, 256, 0, stream>>>(xh, modT[i], mod_b, modbuf[i], KK, npadMod[i], 1);
        break;
      case 3:
        k_conv_gemm<3, 1><<<MTOT / 32, 256, 0, stream>>>(xh, offT[i], off_b, offbuf[i], 2 * KK, npadOff[i], 0);
        k_conv_gemm<3, 1><<<MTOT / 32, 256, 0, stream>>>(xh, modT[i], mod_b, modbuf[i], KK, npadMod[i], 1);
        break;
      default:
        k_conv_gemm<7, 3><<<MTOT / 32, 256, 0, stream>>>(xh, offT[i], off_b, offbuf[i], 2 * KK, npadOff[i], 0);
        k_conv_gemm<7, 3><<<MTOT / 32, 256, 0, stream>>>(xh, modT[i], mod_b, modbuf[i], KK, npadMod[i], 1);
        break;
    }
  }

  // 4. deformable main convs (WMMA + TDM B-panels, templated on k) -> concat
  for (int i = 0; i < 4; ++i) {
    const float* g = (const float*)d_in[1 + 9 * i + 5];
    const float* b = (const float*)d_in[1 + 9 * i + 6];
    const float* m = (const float*)d_in[1 + 9 * i + 7];
    const float* v = (const float*)d_in[1 + 9 * i + 8];
    switch (ks[i]) {
      case 1:
        k_deform_gemm<1, 0><<<MTOT / 32, 256, 0, stream>>>(xh, offbuf[i], modbuf[i], wT[i], g, b, m, v, cat, i * 256);
        break;
      case 3:
        k_deform_gemm<3, 1><<<MTOT / 32, 256, 0, stream>>>(xh, offbuf[i], modbuf[i], wT[i], g, b, m, v, cat, i * 256);
        break;
      default:
        k_deform_gemm<7, 3><<<MTOT / 32, 256, 0, stream>>>(xh, offbuf[i], modbuf[i], wT[i], g, b, m, v, cat, i * 256);
        break;
    }
  }

  // 5. pool branch -> concat channels [1024,1280)
  k_pool_mean<<<BATCH * CINCH, 256, 0, stream>>>(x, xmean);
  k_pool_conv<<<(BATCH * PLANES + 255) / 256, 256, 0, stream>>>(
      xmean, (const float*)d_in[37], (const float*)d_in[38], (const float*)d_in[39],
      (const float*)d_in[40], (const float*)d_in[41], pout);
  k_pool_bcast<<<(MTOT * PLANES + 255) / 256, 256, 0, stream>>>(pout, cat);

  // 6. fuse 1x1 conv (WMMA GEMM) -> f32 NCHW output
  k_fuse_gemm<<<MTOT / 32, 256, 0, stream>>>(cat, fuseT,
      (const float*)d_in[43], (const float*)d_in[44], (const float*)d_in[45],
      (const float*)d_in[46], (float*)d_out);
}
